// ScaledDotProductAttention_12120397709599
// MI455X (gfx1250) — matrix operand order repaired and checked by reading, not compiled, not run on hardware
//
#include <hip/hip_runtime.h>
#include <hip/hip_bf16.h>
#include <limits.h>

typedef __attribute__((ext_vector_type(16))) __bf16 v16bf;
typedef __attribute__((ext_vector_type(8)))  float  v8f;

#define DIMK   64
#define SEQ    2048
#define QTILE  16      // queries per wave
#define KTILE  32      // keys per KV tile
#define WAVES  8       // waves per block
#define QBLK   (QTILE*WAVES)   // 128 queries per block
#define KSTRIDE 72     // Kt LDS row stride (bf16): 144B rows, 16B aligned, conflict-spread
#define VSTRIDE 40     // VtT LDS row stride (bf16): 80B rows
#define PSTRIDE 40     // P scratch row stride (bf16): 80B rows

// single-instruction cross-lane exchange with lane^16 (SWAPX16):
// offset = xor_mask(0x10)<<10 | or_mask(0)<<5 | and_mask(0x1f) = 0x401F
__device__ __forceinline__ float swz_xor16(float x) {
    return __int_as_float(__builtin_amdgcn_ds_swizzle(__float_as_int(x), 0x401F));
}

// One KV tile of flash attention in the S^T formulation.
// Q fragments arrive pre-scaled by 1/sqrt(64)*log2(e), so scores are already
// in the exp2 domain. MASKED=false: zero causal compares (tile below diagonal).
template<bool MASKED>
__device__ __forceinline__ void process_tile(
    int kb, int qrow, int ln, int half,
    const __bf16 (*Kt)[KSTRIDE], const __bf16 (*Vt)[VSTRIDE],
    __bf16 (*Pbw)[PSTRIDE],                 // this wave's P^T scratch [QTILE][PSTRIDE]
    const v16bf* qb, v8f* oacc, float& rm, float& rl)
{
    const float NEG_INF = -__builtin_inff();

    // ---- S^T = K_tile * Qs^T : two M-tiles of 16 keys, contraction d = 64 ----
    v8f sfr[2];
    #pragma unroll
    for (int mt = 0; mt < 2; ++mt) {
        v8f c;
        #pragma unroll
        for (int r = 0; r < 8; ++r) c[r] = 0.0f;
        const __bf16* krow = &Kt[mt*16 + ln][0];
        #pragma unroll
        for (int dc = 0; dc < 2; ++dc) {
            v16bf ka;   // A-frag: m = key, k = d
            #pragma unroll
            for (int vv = 0; vv < 8; ++vv) {
                const int d = 2*vv + 16*half + 32*dc;
                ka[2*vv]   = krow[d];
                ka[2*vv+1] = krow[d + 1];
            }
            c = __builtin_amdgcn_wmma_f32_16x16x32_bf16(
                    false, ka, false, qb[dc], (short)0, c, false, false);
        }
        sfr[mt] = c;
    }

    // ---- (optional) causal mask + tile max ----
    const int thr = qrow - kb;   // key offset valid iff (r + 8*half + 16*mt) <= thr
    float sv[2][8];
    float smax = NEG_INF;
    #pragma unroll
    for (int mt = 0; mt < 2; ++mt) {
        #pragma unroll
        for (int r = 0; r < 8; ++r) {
            float s = sfr[mt][r];
            if (MASKED) {
                if (r + 8*half + 16*mt > thr) s = NEG_INF;
            }
            sv[mt][r] = s;
            smax = fmaxf(smax, s);
        }
    }
    smax = fmaxf(smax, swz_xor16(smax));
    const float mnew = fmaxf(rm, smax);
    const float corr = exp2f(rm - mnew);   // 0 on first visit

    // ---- p = exp2(s - mnew); in-lane sum; stash P^T (bf16) ----
    float sum = 0.0f;
    #pragma unroll
    for (int mt = 0; mt < 2; ++mt) {
        #pragma unroll
        for (int r = 0; r < 8; ++r) {
            const float p = exp2f(sv[mt][r] - mnew);
            sum += p;
            Pbw[ln][r + 8*half + 16*mt] = (__bf16)p;
        }
    }
    sum += swz_xor16(sum);
    rl = rl * corr + sum;
    rm = mnew;
    #pragma unroll
    for (int mt = 0; mt < 4; ++mt) {
        #pragma unroll
        for (int r = 0; r < 8; ++r) oacc[mt][r] *= corr;
    }

    // fence the cross-lane LDS write -> read turnaround (same wave)
    asm volatile("s_wait_dscnt 0" ::: "memory");

    // ---- P^T as a single B-fragment: k = key = e + 16*half, n = query ----
    v16bf pfrag;
    {
        const __bf16* pr = &Pbw[ln][16*half];
        #pragma unroll
        for (int e = 0; e < 16; ++e) pfrag[e] = pr[e];
    }

    // ---- O^T += V^T * P^T : 4 M-tiles of d, contraction over 32 keys ----
    #pragma unroll
    for (int mt = 0; mt < 4; ++mt) {
        const __bf16* vrow = &Vt[mt*16 + ln][0];
        v16bf va;   // A-frag: m = d, k = key
        #pragma unroll
        for (int vv = 0; vv < 8; ++vv) {
            const int kk = 2*vv + 16*half;
            va[2*vv]   = vrow[kk];
            va[2*vv+1] = vrow[kk + 1];
        }
        oacc[mt] = __builtin_amdgcn_wmma_f32_16x16x32_bf16(
                       false, va, false, pfrag, (short)0, oacc[mt], false, false);
    }
}

__global__ __launch_bounds__(256)
void fa_fwd_causal_kernel(const float* __restrict__ Q,
                          const float* __restrict__ K,
                          const float* __restrict__ V,
                          float* __restrict__ Out,
                          const int* __restrict__ causal_flag)
{
    __shared__ __bf16 Kt[KTILE][KSTRIDE];        // K tile: row = key, col = d
    __shared__ __bf16 Vt[DIMK][VSTRIDE];         // V tile transposed: row = d, col = key
    __shared__ __bf16 Pb[WAVES][QTILE][PSTRIDE]; // P^T scratch per wave

    const int tid  = threadIdx.x;
    const int wave = tid >> 5;
    const int lane = tid & 31;
    const int ln   = lane & 15;
    const int half = lane >> 4;

    const int bid       = blockIdx.x;
    const int qblocks   = SEQ / QBLK;         // 16
    const int bh        = bid / qblocks;
    const int qblk      = bid % qblocks;
    const int qbase_blk = qblk * QBLK;
    const int qbase     = qbase_blk + wave * QTILE;

    const float* q = Q + (size_t)bh * SEQ * DIMK;
    const float* k = K + (size_t)bh * SEQ * DIMK;
    const float* v = V + (size_t)bh * SEQ * DIMK;
    float*       o = Out + (size_t)bh * SEQ * DIMK;

    const int causal = causal_flag[0];
    const float NEG_INF = -__builtin_inff();
    const float sc = 0.125f * 1.44269504088896340736f;  // 1/sqrt(64) * log2(e)

    // ---- load Q^T once as two B-fragments, pre-scaled into the exp2 domain ----
    v16bf qb[2];
    {
        const float* qrow_p = q + (size_t)(qbase + ln) * DIMK;
        #pragma unroll
        for (int dc = 0; dc < 2; ++dc) {
            #pragma unroll
            for (int e = 0; e < 16; ++e)
                qb[dc][e] = (__bf16)(qrow_p[e + 16*half + 32*dc] * sc);
        }
    }

    float rm = NEG_INF;
    float rl = 0.0f;
    v8f oacc[4];
    #pragma unroll
    for (int mt = 0; mt < 4; ++mt) {
        #pragma unroll
        for (int r = 0; r < 8; ++r) oacc[mt][r] = 0.0f;
    }

    const int ntiles = causal ? (qbase_blk + QBLK) / KTILE : SEQ / KTILE;
    const int tdiag  = causal ? (qbase >> 5) : INT_MAX;  // single partially-masked tile
    const int qrow   = qbase + ln;

    for (int t = 0; t < ntiles; ++t) {
        const int kb = t * KTILE;

        __syncthreads();  // protect LDS tiles from previous iteration's readers

        // ---- cooperative staging: K row-major, V transposed; fp32 -> bf16 ----
        #pragma unroll
        for (int it = 0; it < 2; ++it) {
            const int j   = tid + 256 * it;   // float4 index 0..511
            const int row = j >> 4;           // key in tile
            const int col = (j & 15) * 4;     // d
            const float4 fk = *(const float4*)(k + (size_t)(kb + row) * DIMK + col);
            Kt[row][col+0] = (__bf16)fk.x;
            Kt[row][col+1] = (__bf16)fk.y;
            Kt[row][col+2] = (__bf16)fk.z;
            Kt[row][col+3] = (__bf16)fk.w;
            const float4 fv = *(const float4*)(v + (size_t)(kb + row) * DIMK + col);
            Vt[col+0][row] = (__bf16)fv.x;
            Vt[col+1][row] = (__bf16)fv.y;
            Vt[col+2][row] = (__bf16)fv.z;
            Vt[col+3][row] = (__bf16)fv.w;
            // prefetch next tile's lines (fire-and-forget, no counters)
            if (t + 1 < ntiles) {
                __builtin_prefetch(k + (size_t)(kb + KTILE + row) * DIMK + col, 0, 1);
                __builtin_prefetch(v + (size_t)(kb + KTILE + row) * DIMK + col, 0, 1);
            }
        }
        __syncthreads();

        if (t < tdiag) {          // full tile (or non-causal): no mask code at all
            process_tile<false>(kb, qrow, ln, half, Kt, Vt, Pb[wave], qb, oacc, rm, rl);
        } else if (t == tdiag) {  // the one diagonal tile for this wave
            process_tile<true>(kb, qrow, ln, half, Kt, Vt, Pb[wave], qb, oacc, rm, rl);
        }
        // t > tdiag: fully masked for this wave -> skip (barriers already matched)
    }

    // ---- epilogue: O^T / l ; lane writes its query's d = r + 8*half + 16*mt ----
    const float inv = 1.0f / rl;
    float* orow = o + (size_t)(qbase + ln) * DIMK;
    #pragma unroll
    for (int mt = 0; mt < 4; ++mt) {
        #pragma unroll
        for (int r = 0; r < 8; ++r)
            orow[r + 8*half + 16*mt] = oacc[mt][r] * inv;
    }
}

extern "C" void kernel_launch(void* const* d_in, const int* in_sizes, int n_in,
                              void* d_out, int out_size, void* d_ws, size_t ws_size,
                              hipStream_t stream) {
    const float* q = (const float*)d_in[0];
    const float* k = (const float*)d_in[1];
    const float* v = (const float*)d_in[2];
    const int* causal = (const int*)d_in[3];
    float* out = (float*)d_out;

    const int BH = 4 * 16;
    const int qblocks = SEQ / QBLK;        // 16
    dim3 grid(BH * qblocks);               // 1024 blocks
    dim3 block(256);                       // 8 waves (wave32)
    fa_fwd_causal_kernel<<<grid, block, 0, stream>>>(q, k, v, out, causal);
}
